// SingleStreamBlock_14903536517134
// MI455X (gfx1250) — compile-verified, matrix operand order checked by reading, not run on hardware
//
#include <hip/hip_runtime.h>
#include <math.h>

// ---------------------------------------------------------------------------
// MI455X (gfx1250) implementation. wave32. All GEMMs via v_wmma_f32_16x16x32_bf16.
// Wave tiles sized so every A/B fragment feeds 2 WMMAs; explicit register
// double-buffering gives each fragment-load 8 WMMAs of latency cover.
// ---------------------------------------------------------------------------

typedef __bf16 bf16;
typedef __bf16 v16bf __attribute__((ext_vector_type(16)));
typedef __bf16 v8bf  __attribute__((ext_vector_type(8)));
typedef float  v8f   __attribute__((ext_vector_type(8)));

#define HIDDEN     1024
#define HEADS      16
#define HEAD_DIM   64
#define MLP_HIDDEN 4096
#define BATCH      2
#define SEQ        2048
#define TOKENS     (BATCH * SEQ)              // 4096
#define N1         (3 * HIDDEN + MLP_HIDDEN)  // 7168
#define QKV        (3 * HIDDEN)               // 3072

// ---- WMMA helpers ----------------------------------------------------------
// A-fragment (16x32 bf16, MxK) per ISA layout: lane L (0..15) holds row M=L,
// K = kb..kb+7 in v0..v3 and kb+16..kb+23 in v4..v7, where kb = k0 + (L>=16)*8.
// Works for B-fragments too when the B source is stored transposed ([N][K]).
__device__ __forceinline__ v16bf load_frag(const bf16* __restrict__ base, int ld,
                                           int row0, int k0, int lane) {
  int r  = row0 + (lane & 15);
  int kb = k0 + ((lane >> 4) << 3);
  const bf16* p = base + (size_t)r * (size_t)ld + kb;
  v8bf lo = *reinterpret_cast<const v8bf*>(p);
  v8bf hi = *reinterpret_cast<const v8bf*>(p + 16);
  return __builtin_shufflevector(lo, hi, 0, 1, 2, 3, 4, 5, 6, 7,
                                 8, 9, 10, 11, 12, 13, 14, 15);
}

__device__ __forceinline__ v8f wmma_bf16(v16bf a, v16bf b, v8f c) {
  // (neg_a, A, neg_b, B, c_mod, C, reuse_a, reuse_b)
  return __builtin_amdgcn_wmma_f32_16x16x32_bf16(false, a, false, b, (short)0, c,
                                                 false, false);
}

// C/D layout: lane l, acc[r] -> (m = r + 8*(l>>4), n = l&15)

// ---- kernel 1: mod = silu(vec) @ w_mod + b_mod -----------------------------
__global__ void mod_kernel(const float* __restrict__ vec,
                           const float* __restrict__ w_mod,
                           const float* __restrict__ b_mod,
                           float* __restrict__ modbuf) {
  __shared__ float sv[HIDDEN];
  int b = blockIdx.x / 12;                       // 12 blocks per batch row
  int j = (blockIdx.x % 12) * 256 + threadIdx.x; // 0..3071
  for (int i = threadIdx.x; i < HIDDEN; i += 256) {
    float t = vec[b * HIDDEN + i];
    sv[i] = t / (1.0f + __expf(-t));
  }
  __syncthreads();
  float acc = 0.0f;
  for (int k = 0; k < HIDDEN; ++k)
    acc += sv[k] * w_mod[(size_t)k * QKV + j];
  modbuf[b * QKV + j] = acc + b_mod[j];
}

// ---- block reduce (256 threads = 8 waves) ----------------------------------
__device__ __forceinline__ float block_sum256(float v, float* s) {
  int lane = threadIdx.x & 31, w = threadIdx.x >> 5;
#pragma unroll
  for (int m = 16; m; m >>= 1) v += __shfl_xor(v, m, 32);
  __syncthreads();
  if (lane == 0) s[w] = v;
  __syncthreads();
  float t = 0.0f;
#pragma unroll
  for (int i = 0; i < 8; ++i) t += s[i];
  return t;
}

// ---- kernel 2: x_mod = (1+scale)*LN(x) + shift, cast bf16 ------------------
__global__ void ln_mod_kernel(const float* __restrict__ x,
                              const float* __restrict__ modbuf,
                              bf16* __restrict__ xb) {
  __shared__ float sred[8];
  int row = blockIdx.x;
  int b = row >> 11;
  const float4* xr = reinterpret_cast<const float4*>(x + (size_t)row * HIDDEN);
  float4 f = xr[threadIdx.x];
  float mean = block_sum256(f.x + f.y + f.z + f.w, sred) * (1.0f / HIDDEN);
  float dx = f.x - mean, dy = f.y - mean, dz = f.z - mean, dw = f.w - mean;
  float var = block_sum256(dx * dx + dy * dy + dz * dz + dw * dw, sred) * (1.0f / HIDDEN);
  float rstd = rsqrtf(var + 1e-6f);
  int n = threadIdx.x * 4;
  const float* mb = modbuf + b * QKV;
  float e[4] = {dx, dy, dz, dw};
#pragma unroll
  for (int j = 0; j < 4; ++j) {
    float sh = mb[n + j];
    float sc = mb[HIDDEN + n + j];
    xb[(size_t)row * HIDDEN + n + j] = (bf16)((1.0f + sc) * (e[j] * rstd) + sh);
  }
}

// ---- kernel 3: weight transpose + bf16 cast: src[K][N] -> dst[N][K] --------
__global__ void transpose_cast_kernel(const float* __restrict__ src,
                                      bf16* __restrict__ dst, int K, int N) {
  size_t total = (size_t)K * N;
  size_t stride = (size_t)gridDim.x * 256;
  for (size_t idx = (size_t)blockIdx.x * 256 + threadIdx.x; idx < total; idx += stride) {
    size_t k = idx / (size_t)N, n = idx % (size_t)N;
    dst[n * (size_t)K + k] = (bf16)src[idx];
  }
}

// ---- WMMA GEMM core: wave computes 32x64 tile of A[M,K] @ Bt[N,K]^T --------
// Explicit one-iteration register double buffer: next iteration's 6 fragments
// are in flight while the current 8 WMMAs execute.
__device__ __forceinline__ void gemm_tile2(const bf16* __restrict__ A,
                                           const bf16* __restrict__ Bt, int K,
                                           int m0, int n0, int lane, v8f acc[2][4]) {
  v16bf a0 = load_frag(A, K, m0, 0, lane);
  v16bf a1 = load_frag(A, K, m0 + 16, 0, lane);
  v16bf b0 = load_frag(Bt, K, n0, 0, lane);
  v16bf b1 = load_frag(Bt, K, n0 + 16, 0, lane);
  v16bf b2 = load_frag(Bt, K, n0 + 32, 0, lane);
  v16bf b3 = load_frag(Bt, K, n0 + 48, 0, lane);
#pragma unroll 2
  for (int k = 0; k < K - 32; k += 32) {
    int kn = k + 32;
    v16bf na0 = load_frag(A, K, m0, kn, lane);
    v16bf na1 = load_frag(A, K, m0 + 16, kn, lane);
    v16bf nb0 = load_frag(Bt, K, n0, kn, lane);
    v16bf nb1 = load_frag(Bt, K, n0 + 16, kn, lane);
    v16bf nb2 = load_frag(Bt, K, n0 + 32, kn, lane);
    v16bf nb3 = load_frag(Bt, K, n0 + 48, kn, lane);
    acc[0][0] = wmma_bf16(a0, b0, acc[0][0]);
    acc[1][0] = wmma_bf16(a1, b0, acc[1][0]);
    acc[0][1] = wmma_bf16(a0, b1, acc[0][1]);
    acc[1][1] = wmma_bf16(a1, b1, acc[1][1]);
    acc[0][2] = wmma_bf16(a0, b2, acc[0][2]);
    acc[1][2] = wmma_bf16(a1, b2, acc[1][2]);
    acc[0][3] = wmma_bf16(a0, b3, acc[0][3]);
    acc[1][3] = wmma_bf16(a1, b3, acc[1][3]);
    a0 = na0; a1 = na1; b0 = nb0; b1 = nb1; b2 = nb2; b3 = nb3;
  }
  acc[0][0] = wmma_bf16(a0, b0, acc[0][0]);
  acc[1][0] = wmma_bf16(a1, b0, acc[1][0]);
  acc[0][1] = wmma_bf16(a0, b1, acc[0][1]);
  acc[1][1] = wmma_bf16(a1, b1, acc[1][1]);
  acc[0][2] = wmma_bf16(a0, b2, acc[0][2]);
  acc[1][2] = wmma_bf16(a1, b2, acc[1][2]);
  acc[0][3] = wmma_bf16(a0, b3, acc[0][3]);
  acc[1][3] = wmma_bf16(a1, b3, acc[1][3]);
}

// 256 threads = 8 waves: 2 M-waves x 4 N-waves -> 64M x 256N per workgroup.
#define GEMM_PROLOGUE(Kdim)                                        \
  int lane = threadIdx.x & 31, w = threadIdx.x >> 5;               \
  int m0 = blockIdx.y * 64 + (w >> 2) * 32;                        \
  int n0 = blockIdx.x * 256 + (w & 3) * 64;                        \
  v8f acc[2][4] = {};                                              \
  gemm_tile2(A, Bt, (Kdim), m0, n0, lane, acc);                    \
  int half = lane >> 4, nn = lane & 15;

// h = x_mod @ w1 + b1 ; cols < 3072 -> Hqkv (f32), cols >= 3072 -> Amlp (bf16)
__global__ void gemm_h_kernel(const bf16* __restrict__ A, const bf16* __restrict__ Bt,
                              const float* __restrict__ b1,
                              float* __restrict__ Hqkv, bf16* __restrict__ Amlp) {
  GEMM_PROLOGUE(HIDDEN)
#pragma unroll
  for (int mi = 0; mi < 2; ++mi)
#pragma unroll
    for (int t = 0; t < 4; ++t)
#pragma unroll
      for (int r = 0; r < 8; ++r) {
        int m = m0 + mi * 16 + r + 8 * half;
        int n = n0 + 16 * t + nn;
        float v = acc[mi][t][r] + b1[n];
        if (n < QKV) Hqkv[(size_t)m * QKV + n] = v;
        else Amlp[(size_t)m * MLP_HIDDEN + (n - QKV)] = (bf16)v;
      }
}

// mlp_out = tanh(mlp @ w_mlp + b_mlp) -> A2[:, 1024:2048] (bf16)
__global__ void gemm_mlp_kernel(const bf16* __restrict__ A, const bf16* __restrict__ Bt,
                                const float* __restrict__ bmlp, bf16* __restrict__ A2) {
  GEMM_PROLOGUE(MLP_HIDDEN)
#pragma unroll
  for (int mi = 0; mi < 2; ++mi)
#pragma unroll
    for (int t = 0; t < 4; ++t)
#pragma unroll
      for (int r = 0; r < 8; ++r) {
        int m = m0 + mi * 16 + r + 8 * half;
        int n = n0 + 16 * t + nn;
        A2[(size_t)m * (2 * HIDDEN) + HIDDEN + n] = (bf16)tanhf(acc[mi][t][r] + bmlp[n]);
      }
}

// out = x + gate * (A2 @ w2 + b2)
__global__ void gemm_out_kernel(const bf16* __restrict__ A, const bf16* __restrict__ Bt,
                                const float* __restrict__ b2,
                                const float* __restrict__ x,
                                const float* __restrict__ modbuf,
                                float* __restrict__ out) {
  GEMM_PROLOGUE(2 * HIDDEN)
#pragma unroll
  for (int mi = 0; mi < 2; ++mi)
#pragma unroll
    for (int t = 0; t < 4; ++t)
#pragma unroll
      for (int r = 0; r < 8; ++r) {
        int m = m0 + mi * 16 + r + 8 * half;
        int n = n0 + 16 * t + nn;
        int b = m >> 11;
        float v = acc[mi][t][r] + b2[n];
        float gate = modbuf[b * QKV + 2 * HIDDEN + n];
        out[(size_t)m * HIDDEN + n] = x[(size_t)m * HIDDEN + n] + gate * v;
      }
}

// ---- kernel: per-head RMSNorm + RoPE + bf16 pack ---------------------------
// One wave per (b,h,l). Lane t owns d=t and d=t+32.
__global__ void qkv_kernel(const float* __restrict__ Hqkv, const float* __restrict__ pe,
                           const float* __restrict__ q_scale, const float* __restrict__ k_scale,
                           bf16* __restrict__ Qb, bf16* __restrict__ Kb, bf16* __restrict__ VT) {
  int lane = threadIdx.x & 31, w = threadIdx.x >> 5;
  int gid = blockIdx.x * 8 + w;          // 0 .. B*H*SEQ-1
  int l = gid & (SEQ - 1);
  int bh = gid >> 11;
  int b = bh >> 4, h = bh & 15;
  const float* hp = Hqkv + ((size_t)(b * SEQ + l)) * QKV + h * HEAD_DIM;

  float q1 = hp[lane], q2 = hp[lane + 32];
  float k1 = hp[HIDDEN + lane], k2 = hp[HIDDEN + lane + 32];
  float v1 = hp[2 * HIDDEN + lane], v2 = hp[2 * HIDDEN + lane + 32];

  float sq = q1 * q1 + q2 * q2, sk = k1 * k1 + k2 * k2;
#pragma unroll
  for (int m = 16; m; m >>= 1) { sq += __shfl_xor(sq, m, 32); sk += __shfl_xor(sk, m, 32); }
  float rq = rsqrtf(sq * (1.0f / HEAD_DIM) + 1e-6f);
  float rk = rsqrtf(sk * (1.0f / HEAD_DIM) + 1e-6f);
  q1 *= rq * q_scale[lane]; q2 *= rq * q_scale[lane + 32];
  k1 *= rk * k_scale[lane]; k2 *= rk * k_scale[lane + 32];

  // RoPE: out[2i+r] = pe[l,i,r,0]*t[2i] + pe[l,i,r,1]*t[2i+1]
  const float* pl = pe + (size_t)l * 128;  // (32,2,2) per position
  int odd = lane & 1;
  float q1p = __shfl_xor(q1, 1, 32), q2p = __shfl_xor(q2, 1, 32);
  float k1p = __shfl_xor(k1, 1, 32), k2p = __shfl_xor(k2, 1, 32);
  float qe1 = odd ? q1p : q1, qo1 = odd ? q1 : q1p;
  float qe2 = odd ? q2p : q2, qo2 = odd ? q2 : q2p;
  float ke1 = odd ? k1p : k1, ko1 = odd ? k1 : k1p;
  float ke2 = odd ? k2p : k2, ko2 = odd ? k2 : k2p;
  int i1 = lane >> 1, i2 = 16 + (lane >> 1);
  float c1 = pl[(i1 * 2 + odd) * 2 + 0], s1 = pl[(i1 * 2 + odd) * 2 + 1];
  float c2 = pl[(i2 * 2 + odd) * 2 + 0], s2 = pl[(i2 * 2 + odd) * 2 + 1];

  size_t qkrow = ((size_t)bh * SEQ + l) * HEAD_DIM;
  Qb[qkrow + lane]      = (bf16)(c1 * qe1 + s1 * qo1);
  Qb[qkrow + lane + 32] = (bf16)(c2 * qe2 + s2 * qo2);
  Kb[qkrow + lane]      = (bf16)(c1 * ke1 + s1 * ko1);
  Kb[qkrow + lane + 32] = (bf16)(c2 * ke2 + s2 * ko2);
  size_t vbase = (size_t)bh * HEAD_DIM * SEQ;
  VT[vbase + (size_t)lane * SEQ + l]        = (bf16)v1;
  VT[vbase + (size_t)(lane + 32) * SEQ + l] = (bf16)v2;
}

// ---- flash attention: 4 waves / block, wave = 32 query rows ----------------
// K and V fragments are each consumed by two WMMAs (two query m-tiles).
// V fragments are issued before the softmax VALU pass so their latency is
// hidden behind ~130 VALU instructions.
__global__ void attn_kernel(const bf16* __restrict__ Qb, const bf16* __restrict__ Kb,
                            const bf16* __restrict__ VT, bf16* __restrict__ A2) {
  __shared__ __align__(16) bf16 pbuf[4][32 * 32];
  int lane = threadIdx.x & 31, w = threadIdx.x >> 5;
  int bh = blockIdx.y, b = bh >> 4, h = bh & 15;
  int q0 = blockIdx.x * 128 + w * 32;
  const bf16* Q  = Qb + (size_t)bh * SEQ * HEAD_DIM;
  const bf16* Kp = Kb + (size_t)bh * SEQ * HEAD_DIM;
  const bf16* Vt = VT + (size_t)bh * HEAD_DIM * SEQ;
  bf16* pw = pbuf[w];

  v16bf qa00 = load_frag(Q, HEAD_DIM, q0, 0, lane);
  v16bf qa01 = load_frag(Q, HEAD_DIM, q0, 32, lane);
  v16bf qa10 = load_frag(Q, HEAD_DIM, q0 + 16, 0, lane);
  v16bf qa11 = load_frag(Q, HEAD_DIM, q0 + 16, 32, lane);

  v8f o[2][4] = {};
  float rm[2][8], rs[2][8];
#pragma unroll
  for (int mi = 0; mi < 2; ++mi)
#pragma unroll
    for (int r = 0; r < 8; ++r) { rm[mi][r] = -1e30f; rs[mi][r] = 0.0f; }
  const float sc = 0.125f;  // 1/sqrt(64)
  int half = lane >> 4, nn = lane & 15;

  for (int kb = 0; kb < SEQ; kb += 32) {
    v16bf kf00 = load_frag(Kp, HEAD_DIM, kb, 0, lane);
    v16bf kf01 = load_frag(Kp, HEAD_DIM, kb, 32, lane);
    v16bf kf10 = load_frag(Kp, HEAD_DIM, kb + 16, 0, lane);
    v16bf kf11 = load_frag(Kp, HEAD_DIM, kb + 16, 32, lane);
    v8f s[2][2] = {};
    s[0][0] = wmma_bf16(qa00, kf00, s[0][0]);
    s[0][0] = wmma_bf16(qa01, kf01, s[0][0]);
    s[0][1] = wmma_bf16(qa00, kf10, s[0][1]);
    s[0][1] = wmma_bf16(qa01, kf11, s[0][1]);
    s[1][0] = wmma_bf16(qa10, kf00, s[1][0]);
    s[1][0] = wmma_bf16(qa11, kf01, s[1][0]);
    s[1][1] = wmma_bf16(qa10, kf10, s[1][1]);
    s[1][1] = wmma_bf16(qa11, kf11, s[1][1]);

    // Issue V fragments now; consumed after the softmax pass below.
    v16bf vf0 = load_frag(Vt, SEQ, 0,  kb, lane);
    v16bf vf1 = load_frag(Vt, SEQ, 16, kb, lane);
    v16bf vf2 = load_frag(Vt, SEQ, 32, kb, lane);
    v16bf vf3 = load_frag(Vt, SEQ, 48, kb, lane);

#pragma unroll
    for (int mi = 0; mi < 2; ++mi)
#pragma unroll
      for (int r = 0; r < 8; ++r) {
        float a = s[mi][0][r] * sc, c = s[mi][1][r] * sc;
        float tm = fmaxf(a, c);
#pragma unroll
        for (int m = 8; m; m >>= 1) tm = fmaxf(tm, __shfl_xor(tm, m, 32));
        float mn = fmaxf(rm[mi][r], tm);
        float alpha = __expf(rm[mi][r] - mn);
        rm[mi][r] = mn;
        float e0 = __expf(a - mn), e1 = __expf(c - mn);
        float ls = e0 + e1;
#pragma unroll
        for (int m = 8; m; m >>= 1) ls += __shfl_xor(ls, m, 32);
        rs[mi][r] = rs[mi][r] * alpha + ls;
        o[mi][0][r] *= alpha; o[mi][1][r] *= alpha;
        o[mi][2][r] *= alpha; o[mi][3][r] *= alpha;
        int mrow = mi * 16 + r + 8 * half;
        pw[mrow * 32 + nn]      = (bf16)e0;
        pw[mrow * 32 + 16 + nn] = (bf16)e1;
      }
    asm volatile("s_wait_dscnt 0" ::: "memory");
    v16bf pa0 = load_frag(pw, 32, 0, 0, lane);
    v16bf pa1 = load_frag(pw, 32, 16, 0, lane);
    o[0][0] = wmma_bf16(pa0, vf0, o[0][0]);
    o[1][0] = wmma_bf16(pa1, vf0, o[1][0]);
    o[0][1] = wmma_bf16(pa0, vf1, o[0][1]);
    o[1][1] = wmma_bf16(pa1, vf1, o[1][1]);
    o[0][2] = wmma_bf16(pa0, vf2, o[0][2]);
    o[1][2] = wmma_bf16(pa1, vf2, o[1][2]);
    o[0][3] = wmma_bf16(pa0, vf3, o[0][3]);
    o[1][3] = wmma_bf16(pa1, vf3, o[1][3]);
  }

#pragma unroll
  for (int mi = 0; mi < 2; ++mi)
#pragma unroll
    for (int r = 0; r < 8; ++r) {
      float inv = 1.0f / rs[mi][r];
      int l = q0 + mi * 16 + r + 8 * half;
      size_t base = ((size_t)(b * SEQ + l)) * (2 * HIDDEN) + h * HEAD_DIM + nn;
      A2[base + 0]  = (bf16)(o[mi][0][r] * inv);
      A2[base + 16] = (bf16)(o[mi][1][r] * inv);
      A2[base + 32] = (bf16)(o[mi][2][r] * inv);
      A2[base + 48] = (bf16)(o[mi][3][r] * inv);
    }
}

// ---------------------------------------------------------------------------
extern "C" void kernel_launch(void* const* d_in, const int* in_sizes, int n_in,
                              void* d_out, int out_size, void* d_ws, size_t ws_size,
                              hipStream_t stream) {
  (void)in_sizes; (void)n_in; (void)out_size; (void)ws_size;
  const float* x       = (const float*)d_in[0];
  const float* vec     = (const float*)d_in[1];
  const float* pe      = (const float*)d_in[2];
  const float* w_mod   = (const float*)d_in[3];
  const float* b_mod   = (const float*)d_in[4];
  const float* w1      = (const float*)d_in[5];
  const float* b1      = (const float*)d_in[6];
  const float* w_mlp   = (const float*)d_in[7];
  const float* b_mlp   = (const float*)d_in[8];
  const float* w2      = (const float*)d_in[9];
  const float* b2      = (const float*)d_in[10];
  const float* q_scale = (const float*)d_in[11];
  const float* k_scale = (const float*)d_in[12];
  float* out = (float*)d_out;

  char* p = (char*)d_ws;
  auto alloc = [&](size_t bytes) -> char* {
    char* r = p; p += (bytes + 255) & ~(size_t)255; return r;
  };
  float* modbuf = (float*)alloc((size_t)BATCH * QKV * 4);
  bf16*  xb     = (bf16*) alloc((size_t)TOKENS * HIDDEN * 2);
  bf16*  w1T    = (bf16*) alloc((size_t)N1 * HIDDEN * 2);
  bf16*  wmlpT  = (bf16*) alloc((size_t)HIDDEN * MLP_HIDDEN * 2);
  bf16*  w2T    = (bf16*) alloc((size_t)HIDDEN * 2 * HIDDEN * 2);
  float* Hqkv   = (float*)alloc((size_t)TOKENS * QKV * 4);
  bf16*  Amlp   = (bf16*) alloc((size_t)TOKENS * MLP_HIDDEN * 2);
  bf16*  Qb     = (bf16*) alloc((size_t)BATCH * HEADS * SEQ * HEAD_DIM * 2);
  bf16*  Kb     = (bf16*) alloc((size_t)BATCH * HEADS * SEQ * HEAD_DIM * 2);
  bf16*  VT     = (bf16*) alloc((size_t)BATCH * HEADS * SEQ * HEAD_DIM * 2);
  bf16*  A2     = (bf16*) alloc((size_t)TOKENS * 2 * HIDDEN * 2);

  mod_kernel<<<24, 256, 0, stream>>>(vec, w_mod, b_mod, modbuf);
  ln_mod_kernel<<<TOKENS, 256, 0, stream>>>(x, modbuf, xb);
  transpose_cast_kernel<<<4096, 256, 0, stream>>>(w1, w1T, HIDDEN, N1);
  transpose_cast_kernel<<<4096, 256, 0, stream>>>(w_mlp, wmlpT, MLP_HIDDEN, HIDDEN);
  transpose_cast_kernel<<<2048, 256, 0, stream>>>(w2, w2T, 2 * HIDDEN, HIDDEN);
  gemm_h_kernel<<<dim3(N1 / 256, TOKENS / 64), 256, 0, stream>>>(xb, w1T, b1, Hqkv, Amlp);
  qkv_kernel<<<(BATCH * HEADS * SEQ) / 8, 256, 0, stream>>>(Hqkv, pe, q_scale, k_scale,
                                                            Qb, Kb, VT);
  attn_kernel<<<dim3(SEQ / 128, BATCH * HEADS), 128, 0, stream>>>(Qb, Kb, VT, A2);
  gemm_mlp_kernel<<<dim3(HIDDEN / 256, TOKENS / 64), 256, 0, stream>>>(Amlp, wmlpT, b_mlp, A2);
  gemm_out_kernel<<<dim3(HIDDEN / 256, TOKENS / 64), 256, 0, stream>>>(A2, w2T, b2, x,
                                                                       modbuf, out);
}
